// GATInductive_2499670966451
// MI455X (gfx1250) — matrix operand order repaired and checked by reading, not compiled, not run on hardware
//
#include <hip/hip_runtime.h>
#include <hip/hip_bf16.h>
#include <math.h>

#define NN 100000            // nodes
#define NE 1600000           // input edges
#define ET (NE + NN)         // edges + self loops
#define NEG_SLOPE 0.2f
#define BN_EPS 1e-5f

typedef __attribute__((ext_vector_type(16))) _Float16 v16h;
typedef __attribute__((ext_vector_type(8)))  float    v8f;

static __device__ __forceinline__ unsigned encF(float f) {
  unsigned u = __float_as_uint(f);
  return (u & 0x80000000u) ? ~u : (u | 0x80000000u);
}
static __device__ __forceinline__ float decF(unsigned u) {
  unsigned v = (u & 0x80000000u) ? (u & 0x7FFFFFFFu) : ~u;
  return __uint_as_float(v);
}
static __device__ __forceinline__ float lrelu(float x) {
  return x > 0.f ? x : NEG_SLOPE * x;
}
static __device__ __forceinline__ void edge_sd(const int* __restrict__ ei, int e,
                                               int& s, int& d) {
  if (e < NE) { s = ei[e]; d = ei[NE + e]; }
  else        { s = e - NE; d = e - NE; }
}

// ---------------- small utility kernels ----------------
__global__ void k_zero_f32(float* __restrict__ p, int n) {
  int i = blockIdx.x * blockDim.x + threadIdx.x;
  if (i < n) p[i] = 0.f;
}
__global__ void k_fill_u32(unsigned* __restrict__ p, unsigned v, int n) {
  int i = blockIdx.x * blockDim.x + threadIdx.x;
  if (i < n) p[i] = v;
}

// ---------------- B pre-pack into WMMA lane layout ----------------
// BP[((t16*KS + ks)*32 + lane)*16 + j] = f16(W[(ks*32 + (lane>>4)*16 + j) * N + t16*16 + (lane&15)])
// zero-padded beyond N. Each GEMM lane then reads its 16 halves contiguously (32B).
__global__ void k_pack_b(const float* __restrict__ W, _Float16* __restrict__ BP,
                         int K, int N, int T16) {
  int t = blockIdx.x * blockDim.x + threadIdx.x;
  int KS = K >> 5;
  int total = T16 * KS * 32 * 16;
  if (t >= total) return;
  int j    = t & 15;
  int lane = (t >> 4) & 31;
  int ks   = (t >> 9) % KS;
  int t16  = t / (KS << 9);
  int row = ks * 32 + (lane >> 4) * 16 + j;   // K index
  int col = t16 * 16 + (lane & 15);           // N index
  BP[t] = (col < N) ? (_Float16)W[(size_t)row * N + col] : (_Float16)0.f;
}

// ---------------- WMMA GEMM: C[M,N] = A[M,K](f32) * W(K,N via BP pack) ----------------
// One wave computes a 16x32 tile (two 16x16 WMMA tiles sharing the A fragment).
// K must be a multiple of 32. Branchless inner loop; pointer-increment addressing.
__global__ void k_wmma_gemm(const float* __restrict__ A, const _Float16* __restrict__ BP,
                            float* __restrict__ C, int M, int N, int K) {
  const int lane = threadIdx.x;        // 0..31, full wave, EXEC all ones at WMMA
  const int m0 = blockIdx.x * 16;
  const int n0 = blockIdx.y * 32;
  const int l15 = lane & 15;
  const int hf  = lane >> 4;
  const int KS  = K >> 5;
  const int arow = min(m0 + l15, M - 1);
  const float* ap = A + (size_t)arow * K + hf * 16;           // 32B aligned
  const v16h* bp0 = (const v16h*)BP + (size_t)(2 * blockIdx.y) * KS * 32 + lane;
  const v16h* bp1 = bp0 + (size_t)KS * 32;
  v8f acc0 = {}, acc1 = {};
  for (int ks = 0; ks < KS; ++ks, ap += 32, bp0 += 32, bp1 += 32) {
    if (ks + 1 < KS) __builtin_prefetch(ap + 32, 0, 3);       // global_prefetch_b8
    float4 q0 = ((const float4*)ap)[0];
    float4 q1 = ((const float4*)ap)[1];
    float4 q2 = ((const float4*)(ap + 8))[0];
    float4 q3 = ((const float4*)(ap + 8))[1];
    v16h a;
    a[0]  = (_Float16)q0.x; a[1]  = (_Float16)q0.y;
    a[2]  = (_Float16)q0.z; a[3]  = (_Float16)q0.w;
    a[4]  = (_Float16)q1.x; a[5]  = (_Float16)q1.y;
    a[6]  = (_Float16)q1.z; a[7]  = (_Float16)q1.w;
    a[8]  = (_Float16)q2.x; a[9]  = (_Float16)q2.y;
    a[10] = (_Float16)q2.z; a[11] = (_Float16)q2.w;
    a[12] = (_Float16)q3.x; a[13] = (_Float16)q3.y;
    a[14] = (_Float16)q3.z; a[15] = (_Float16)q3.w;
    v16h b0 = *bp0;
    v16h b1 = *bp1;
    acc0 = __builtin_amdgcn_wmma_f32_16x16x32_f16(false, a, false, b0,
                                                  (short)0, acc0, false, false);
    acc1 = __builtin_amdgcn_wmma_f32_16x16x32_f16(false, a, false, b1,
                                                  (short)0, acc1, false, false);
  }
  const int ncol0 = n0 + l15;
  const int ncol1 = ncol0 + 16;
#pragma unroll
  for (int r = 0; r < 8; ++r) {
    int row = m0 + hf * 8 + r;
    if (row < M) {
      float* crow = C + (size_t)row * N;
      if (ncol0 < N) crow[ncol0] = acc0[r];
      if (ncol1 < N) crow[ncol1] = acc1[r];
    }
  }
}

// ---------------- per-node attention projections ----------------
__global__ void k_att(const float* __restrict__ Hf, const float* __restrict__ as_,
                      const float* __restrict__ ad_, float* __restrict__ a_s,
                      float* __restrict__ a_d, int F, int H, int C) {
  int n = blockIdx.x * blockDim.x + threadIdx.x;
  if (n >= NN) return;
  const float* hr = Hf + (size_t)n * F;
  for (int h = 0; h < H; ++h) {
    float ss = 0.f, sd = 0.f;
    for (int c = 0; c < C; ++c) {
      float v = hr[h * C + c];
      ss += v * as_[h * C + c];
      sd += v * ad_[h * C + c];
    }
    a_s[n * H + h] = ss;
    a_d[n * H + h] = sd;
  }
}

// ---------------- edge passes ----------------
__global__ void k_edge_max(const int* __restrict__ ei, const float* __restrict__ a_s,
                           const float* __restrict__ a_d, unsigned* __restrict__ m, int H) {
  int e = blockIdx.x * blockDim.x + threadIdx.x;
  if (e >= ET) return;
  int s, d; edge_sd(ei, e, s, d);
  for (int h = 0; h < H; ++h) {
    float ev = lrelu(a_s[s * H + h] + a_d[d * H + h]);
    atomicMax(&m[d * H + h], encF(ev));
  }
}

__global__ void k_m_fin(unsigned* __restrict__ mu, float* __restrict__ mf, int n) {
  int i = blockIdx.x * blockDim.x + threadIdx.x;
  if (i >= n) return;
  float v = decF(mu[i]);
  mf[i] = __builtin_isfinite(v) ? v : 0.f;
}

__global__ void k_edge_p(const int* __restrict__ ei, const float* __restrict__ a_s,
                         const float* __restrict__ a_d, const float* __restrict__ m,
                         float* __restrict__ denom, float* __restrict__ alpha, int H) {
  int e = blockIdx.x * blockDim.x + threadIdx.x;
  if (e >= ET) return;
  int s, d; edge_sd(ei, e, s, d);
  for (int h = 0; h < H; ++h) {
    float ev = lrelu(a_s[s * H + h] + a_d[d * H + h]);
    float p  = expf(ev - m[d * H + h]);
    alpha[(size_t)e * H + h] = p;
    atomicAdd(&denom[d * H + h], p);
  }
}

__global__ void k_edge_norm(const int* __restrict__ ei, const float* __restrict__ denom,
                            float* __restrict__ alpha, int H) {
  int t = blockIdx.x * blockDim.x + threadIdx.x;
  if (t >= ET * H) return;
  int e = t / H, h = t - e * H;
  int d = (e < NE) ? ei[NE + e] : e - NE;
  alpha[t] = alpha[t] / (denom[d * H + h] + 1e-16f);
}

// one thread per (edge, feature): coalesced gather of H[src] row + atomic scatter to out[dst]
__global__ void k_edge_aggr(const int* __restrict__ ei, const float* __restrict__ Hf,
                            const float* __restrict__ alpha, float* __restrict__ out,
                            int F, int H, int C) {
  long long t = (long long)blockIdx.x * blockDim.x + threadIdx.x;
  long long total = (long long)ET * F;
  if (t >= total) return;
  int e = (int)(t / F);
  int f = (int)(t - (long long)e * F);
  int s, d; edge_sd(ei, e, s, d);
  int h = f / C;
  float v = Hf[(size_t)s * F + f] * alpha[(size_t)e * H + h];
  atomicAdd(&out[(size_t)d * F + f], v);
}

// ---------------- post-processing ----------------
__global__ void k_bn_elu(const float* __restrict__ agg, const float* __restrict__ b,
                         const float* __restrict__ g, const float* __restrict__ beta,
                         const float* __restrict__ rm, const float* __restrict__ rv,
                         float* __restrict__ out, int F) {
  int t = blockIdx.x * blockDim.x + threadIdx.x;
  if (t >= NN * F) return;
  int f = t % F;
  float v = agg[t] + b[f];
  v = (v - rm[f]) * rsqrtf(rv[f] + BN_EPS) * g[f] + beta[f];
  out[t] = v > 0.f ? v : expm1f(v);
}

__global__ void k_logsoftmax(const float* __restrict__ agg, const float* __restrict__ b,
                             float* __restrict__ out) {
  int n = blockIdx.x * blockDim.x + threadIdx.x;
  if (n >= NN) return;
  float vals[40];
  float mx = -INFINITY;
  for (int i = 0; i < 40; ++i) {
    vals[i] = agg[(size_t)n * 40 + i] + b[i];
    mx = fmaxf(mx, vals[i]);
  }
  float ssum = 0.f;
  for (int i = 0; i < 40; ++i) ssum += expf(vals[i] - mx);
  float l = logf(ssum);
  for (int i = 0; i < 40; ++i) out[(size_t)n * 40 + i] = vals[i] - mx - l;
}

// ---------------- host orchestration ----------------
extern "C" void kernel_launch(void* const* d_in, const int* in_sizes, int n_in,
                              void* d_out, int out_size, void* d_ws, size_t ws_size,
                              hipStream_t stream) {
  (void)in_sizes; (void)n_in; (void)out_size; (void)ws_size;
  const float* x   = (const float*)d_in[0];
  const int*   ei  = (const int*)d_in[1];
  const float* W1  = (const float*)d_in[2];
  const float* as1 = (const float*)d_in[3];
  const float* ad1 = (const float*)d_in[4];
  const float* b1  = (const float*)d_in[5];
  const float* g1  = (const float*)d_in[6];
  const float* be1 = (const float*)d_in[7];
  const float* rm1 = (const float*)d_in[8];
  const float* rv1 = (const float*)d_in[9];
  const float* W2  = (const float*)d_in[10];
  const float* as2 = (const float*)d_in[11];
  const float* ad2 = (const float*)d_in[12];
  const float* b2  = (const float*)d_in[13];
  const float* g2  = (const float*)d_in[14];
  const float* be2 = (const float*)d_in[15];
  const float* rm2 = (const float*)d_in[16];
  const float* rv2 = (const float*)d_in[17];
  const float* W3  = (const float*)d_in[18];
  const float* as3 = (const float*)d_in[19];
  const float* ad3 = (const float*)d_in[20];
  const float* b3  = (const float*)d_in[21];
  float* outp = (float*)d_out;

  // workspace carve-up (256B aligned)
  char* ws = (char*)d_ws;
  size_t off = 0;
  auto carve = [&](size_t bytes) -> void* {
    void* p = (void*)(ws + off);
    off += (bytes + 255) & ~(size_t)255;
    return p;
  };
  float*    Hbuf  = (float*)carve((size_t)NN * 128 * 4);   // node features after GEMM
  float*    Outb  = (float*)carve((size_t)NN * 128 * 4);   // aggregation accumulator
  float*    Xbuf  = (float*)carve((size_t)NN * 128 * 4);   // layer input features
  float*    a_s   = (float*)carve((size_t)NN * 4 * 4);
  float*    a_d   = (float*)carve((size_t)NN * 4 * 4);
  unsigned* mbuf  = (unsigned*)carve((size_t)NN * 4 * 4);
  float*    denom = (float*)carve((size_t)NN * 4 * 4);
  float*    alpha = (float*)carve((size_t)ET * 4 * 4);
  // packed B (f16, WMMA lane layout, zero-padded to 32-col multiples)
  _Float16* BP1   = (_Float16*)carve((size_t)8 * 128 * 16 * 2);  // T16=8, K=128
  _Float16* BP2   = (_Float16*)carve((size_t)2 * 128 * 16 * 2);  // T16=2, K=128
  _Float16* BP3   = (_Float16*)carve((size_t)4 *  32 * 16 * 2);  // T16=4, K=32

  const int TB = 256;
  auto nb = [&](long long n) -> unsigned { return (unsigned)((n + TB - 1) / TB); };

  // B pre-pack (tiny)
  k_pack_b<<<nb((long long)8 * 128 * 16), TB, 0, stream>>>(W1, BP1, 128, 128, 8);
  k_pack_b<<<nb((long long)2 * 128 * 16), TB, 0, stream>>>(W2, BP2, 128, 32, 2);
  k_pack_b<<<nb((long long)4 *  32 * 16), TB, 0, stream>>>(W3, BP3, 32, 40, 4);

  auto gat_layer = [&](const float* xin, int Fin, int Fout, int H,
                       const _Float16* BP, const float* asv, const float* adv) {
    int C = Fout / H;
    dim3 gg((NN + 15) / 16, (Fout + 31) / 32);
    k_wmma_gemm<<<gg, 32, 0, stream>>>(xin, BP, Hbuf, NN, Fout, Fin);
    k_att<<<nb(NN), TB, 0, stream>>>(Hbuf, asv, adv, a_s, a_d, Fout, H, C);
    int NH = NN * H;
    k_fill_u32<<<nb(NH), TB, 0, stream>>>(mbuf, 0x007FFFFFu /* enc(-inf) */, NH);
    k_zero_f32<<<nb(NH), TB, 0, stream>>>(denom, NH);
    k_zero_f32<<<nb((long long)NN * Fout), TB, 0, stream>>>(Outb, NN * Fout);
    k_edge_max<<<nb(ET), TB, 0, stream>>>(ei, a_s, a_d, mbuf, H);
    k_m_fin<<<nb(NH), TB, 0, stream>>>(mbuf, (float*)mbuf, NH);
    k_edge_p<<<nb(ET), TB, 0, stream>>>(ei, a_s, a_d, (float*)mbuf, denom, alpha, H);
    k_edge_norm<<<nb((long long)ET * H), TB, 0, stream>>>(ei, denom, alpha, H);
    long long tot = (long long)ET * Fout;
    k_edge_aggr<<<nb(tot), TB, 0, stream>>>(ei, Hbuf, alpha, Outb, Fout, H, C);
  };

  // Layer 1: 128 -> 4 heads x 32 (concat) ; BN + ELU
  gat_layer(x, 128, 128, 4, BP1, as1, ad1);
  k_bn_elu<<<nb((long long)NN * 128), TB, 0, stream>>>(Outb, b1, g1, be1, rm1, rv1, Xbuf, 128);

  // Layer 2: 128 -> 32 (1 head) ; BN + ELU
  gat_layer(Xbuf, 128, 32, 1, BP2, as2, ad2);
  k_bn_elu<<<nb((long long)NN * 32), TB, 0, stream>>>(Outb, b2, g2, be2, rm2, rv2, Xbuf, 32);

  // Layer 3: 32 -> 40 (1 head) ; log_softmax
  gat_layer(Xbuf, 32, 40, 1, BP3, as3, ad3);
  k_logsoftmax<<<nb(NN), TB, 0, stream>>>(Outb, b3, outp);
}
